// TorchEncoderBlock_25434796327449
// MI455X (gfx1250) — matrix operand order repaired and checked by reading, not compiled, not run on hardware
//
#include <hip/hip_runtime.h>
#include <stdint.h>

typedef __bf16 bf16;
typedef __attribute__((ext_vector_type(16))) __bf16 bf16x16;
typedef __attribute__((ext_vector_type(8)))  float  f32x8;

#define SEQ  2048
#define HID  1024
#define NH   16
#define HD   64
#define MLPD 4096

union FragU { uint4 u[2]; bf16x16 v; };

__device__ __forceinline__ bf16x16 load_frag(const bf16* p0, const bf16* p1) {
  FragU f;
  f.u[0] = *(const uint4*)p0;
  f.u[1] = *(const uint4*)p1;
  return f.v;
}

// CDNA5 async DMA: global -> LDS, 16B per lane, tracked by ASYNCcnt.
// LDS operand: low 32 bits of a flat pointer into __shared__ are the DS address.
__device__ __forceinline__ void async_ld_b128(void* lds, const void* g) {
  asm volatile("global_load_async_to_lds_b128 %0, %1, off"
               :: "v"((uint32_t)(size_t)lds), "v"(g)
               : "memory");
}
__device__ __forceinline__ void wait_async0() {
  asm volatile("s_wait_asynccnt 0x0" ::: "memory");
}

// ---------------------------------------------------------------------------
// fp32 -> bf16 weight conversion (grid: n/1024 blocks x 256 thr, 4 elems/thr)
// ---------------------------------------------------------------------------
__global__ __launch_bounds__(256)
void cvt_f32_bf16_kernel(const float* __restrict__ in, bf16* __restrict__ out, int n) {
  int i = (blockIdx.x * 256 + threadIdx.x) * 4;
  if (i < n) {
    float4 v = *(const float4*)(in + i);
    union { bf16 h[4]; uint2 u; } o;
    o.h[0] = (bf16)v.x; o.h[1] = (bf16)v.y; o.h[2] = (bf16)v.z; o.h[3] = (bf16)v.w;
    *(uint2*)(out + i) = o.u;
  }
}

// ---------------------------------------------------------------------------
// LayerNorm: one block per row (1024 cols, 4/thread), bf16 output
// ---------------------------------------------------------------------------
__global__ __launch_bounds__(256)
void layernorm_bf16_kernel(const float* __restrict__ x, const float* __restrict__ g,
                           const float* __restrict__ bb, bf16* __restrict__ out) {
  const int row = blockIdx.x, tid = threadIdx.x, lane = tid & 31, wv = tid >> 5;
  const float4 v = ((const float4*)(x + (size_t)row * HID))[tid];
  float s  = v.x + v.y + v.z + v.w;
  float ss = v.x * v.x + v.y * v.y + v.z * v.z + v.w * v.w;
#pragma unroll
  for (int m = 1; m < 32; m <<= 1) { s += __shfl_xor(s, m, 32); ss += __shfl_xor(ss, m, 32); }
  __shared__ float sbuf[8], ssbuf[8];
  if (lane == 0) { sbuf[wv] = s; ssbuf[wv] = ss; }
  __syncthreads();
  s = 0.f; ss = 0.f;
#pragma unroll
  for (int i = 0; i < 8; ++i) { s += sbuf[i]; ss += ssbuf[i]; }
  const float mean = s * (1.0f / HID);
  const float inv  = rsqrtf(ss * (1.0f / HID) - mean * mean + 1e-5f);
  const float4 gw = ((const float4*)g)[tid];
  const float4 bw = ((const float4*)bb)[tid];
  union { bf16 h[4]; uint2 u; } o;
  o.h[0] = (bf16)((v.x - mean) * inv * gw.x + bw.x);
  o.h[1] = (bf16)((v.y - mean) * inv * gw.y + bw.y);
  o.h[2] = (bf16)((v.z - mean) * inv * gw.z + bw.z);
  o.h[3] = (bf16)((v.w - mean) * inv * gw.w + bw.w);
  *(uint2*)(out + (size_t)row * HID + tid * 4) = o.u;
}

// ---------------------------------------------------------------------------
// GEMM: out[M,N] = A[M,K](bf16) @ W[N,K]^T(bf16) + bias, optional epilogues
//   mode 0: store fp32 | mode 1: tanh-GELU -> bf16 | mode 2: += resid -> fp32
// Block tile 128x128, BK=32. 8 waves in 4(M) x 2(N); wave tile 32x64.
// Double-buffered LDS fed by GLOBAL_LOAD_ASYNC_TO_LDS_B128, 1 barrier/K-step.
// ---------------------------------------------------------------------------
__global__ __launch_bounds__(256)
void gemm_bf16_kernel(const bf16* __restrict__ A, const bf16* __restrict__ W,
                      const float* __restrict__ bias, const float* __restrict__ resid,
                      float* __restrict__ outF, bf16* __restrict__ outB,
                      int M, int N, int K, int mode) {
  __shared__ __align__(16) bf16 As[2][128 * 32];
  __shared__ __align__(16) bf16 Bs[2][128 * 32];
  const int tid = threadIdx.x, lane = tid & 31, wv = tid >> 5;
  const int wave_m = wv & 3, wave_n = wv >> 2;
  const int bm = blockIdx.y * 128, bn = blockIdx.x * 128;
  const int mrow = lane & 15, hi = lane >> 4;
  f32x8 acc[2][4] = {};

  const int aidx = tid * 16, ar = aidx >> 5, ac = aidx & 31;  // 16 halves/thread
  const bf16* Ag = A + (size_t)(bm + ar) * K + ac;
  const bf16* Wg = W + (size_t)(bn + ar) * K + ac;

  // preload K-tile 0 (async DMA straight into LDS)
  async_ld_b128(&As[0][aidx], Ag);     async_ld_b128(&As[0][aidx + 8], Ag + 8);
  async_ld_b128(&Bs[0][aidx], Wg);     async_ld_b128(&Bs[0][aidx + 8], Wg + 8);
  wait_async0();
  __syncthreads();

  const int nk = K >> 5;
  for (int kt = 0; kt < nk; ++kt) {
    const int cur = kt & 1, nxt = cur ^ 1;
    const bool more = (kt + 1) < nk;
    if (more) {  // DMA next tile into the other buffer; overlaps the WMMAs
      const bf16* Agn = Ag + (kt + 1) * 32;
      const bf16* Wgn = Wg + (kt + 1) * 32;
      async_ld_b128(&As[nxt][aidx], Agn);  async_ld_b128(&As[nxt][aidx + 8], Agn + 8);
      async_ld_b128(&Bs[nxt][aidx], Wgn);  async_ld_b128(&Bs[nxt][aidx + 8], Wgn + 8);
    }
    // load all fragments first, then back-to-back WMMAs
    bf16x16 a[2], b[4];
#pragma unroll
    for (int i = 0; i < 2; ++i) {
      const int row = (wave_m * 32 + i * 16 + mrow) * 32;
      a[i] = load_frag(&As[cur][row + hi * 8], &As[cur][row + 16 + hi * 8]);
    }
#pragma unroll
    for (int t = 0; t < 4; ++t) {
      const int row = (wave_n * 64 + t * 16 + mrow) * 32;
      b[t] = load_frag(&Bs[cur][row + hi * 8], &Bs[cur][row + 16 + hi * 8]);
    }
#pragma unroll
    for (int i = 0; i < 2; ++i)
#pragma unroll
      for (int t = 0; t < 4; ++t)
        acc[i][t] = __builtin_amdgcn_wmma_f32_16x16x32_bf16(
            false, a[i], false, b[t], (short)0, acc[i][t], false, false);
    if (more) wait_async0();  // own DMAs landed; barrier makes them block-visible
    __syncthreads();
  }
  // C layout: VGPR r -> row r (lanes 0-15) / r+8 (lanes 16-31), col = lane%16
#pragma unroll
  for (int i = 0; i < 2; ++i) {
    const int row0 = bm + wave_m * 32 + i * 16 + hi * 8;
#pragma unroll
    for (int t = 0; t < 4; ++t) {
      const int col = bn + wave_n * 64 + t * 16 + mrow;
      const float bv = bias[col];
#pragma unroll
      for (int r = 0; r < 8; ++r) {
        float v = acc[i][t][r] + bv;
        const size_t oi = (size_t)(row0 + r) * N + col;
        if (mode == 1) {
          float u = 0.7978845608028654f * (v + 0.044715f * v * v * v);
          v = 0.5f * v * (1.0f + tanhf(u));
          outB[oi] = (bf16)v;
        } else if (mode == 2) {
          outF[oi] = v + resid[oi];
        } else {
          outF[oi] = v;
        }
      }
    }
  }
}

// ---------------------------------------------------------------------------
// RoPE + head split. qkv fp32 [S][3][NH][HD] -> q/k bf16 [h][s][d] (rotated),
// v bf16 transposed [h][d][s]. One thread per (s,h,pair).
// ---------------------------------------------------------------------------
__global__ __launch_bounds__(256)
void rope_split_kernel(const float* __restrict__ qkv, const float* __restrict__ ct,
                       const float* __restrict__ st, bf16* __restrict__ qg,
                       bf16* __restrict__ kg, bf16* __restrict__ vg) {
  const int gid = blockIdx.x * 256 + threadIdx.x;
  const int i = gid & 31;          // rotation pair index (d = 2i, 2i+1)
  const int h = (gid >> 5) & 15;
  const int s = gid >> 9;
  const float c = ct[s * 32 + i], sn = st[s * 32 + i];
  const float* base = qkv + (size_t)s * 3072 + h * 64 + 2 * i;
  const size_t qo = ((size_t)h * SEQ + s) * HD + 2 * i;
  float re = base[0], im = base[1];
  qg[qo]     = (bf16)(re * c - im * sn);
  qg[qo + 1] = (bf16)(re * sn + im * c);
  re = base[1024]; im = base[1025];
  kg[qo]     = (bf16)(re * c - im * sn);
  kg[qo + 1] = (bf16)(re * sn + im * c);
  const float v0 = base[2048], v1 = base[2049];
  vg[((size_t)h * HD + 2 * i)     * SEQ + s] = (bf16)v0;
  vg[((size_t)h * HD + 2 * i + 1) * SEQ + s] = (bf16)v1;
}

// ---------------------------------------------------------------------------
// Flash attention: grid (SEQ/128, NH), 8 waves, wave owns 16 query rows.
// scores = (Q.K^T)/8 + (same_segment ? 1 : 0)  (additive bias, NOT -inf mask)
// K and V^T tiles are block-shared: async-DMA'd into double-buffered LDS
// (1 barrier/tile). Online softmax; P re-layouts C->A through a per-wave LDS
// patch (DS ops are in-order within a wave).
// ---------------------------------------------------------------------------
__global__ __launch_bounds__(256)
void flash_attn_kernel(const bf16* __restrict__ qg, const bf16* __restrict__ kg,
                       const bf16* __restrict__ vg, bf16* __restrict__ og,
                       const int* __restrict__ offs) {
  __shared__ __align__(16) bf16 Ks[2][64 * 64];   // [key][d]  8 KB each
  __shared__ __align__(16) bf16 Vs[2][64 * 64];   // [d][key]  8 KB each
  __shared__ __align__(16) bf16 Ps[8 * 16 * 64];  // per-wave 16x64 P patch
  const int tid = threadIdx.x, lane = tid & 31, wv = tid >> 5;
  const int h = blockIdx.y;
  const int q0 = blockIdx.x * 128 + wv * 16;
  const int mrow = lane & 15, hi = lane >> 4;
  const int o1 = offs[1], o2 = offs[2], o3 = offs[3];

  // staging indices: 16 halves / thread into a 64x64 tile (row stride 64)
  const int sidx = tid * 16, srow = sidx >> 6, scol = sidx & 63;
  const bf16* Kg0 = kg + ((size_t)h * SEQ + srow) * HD + scol;   // [key][d]
  const bf16* Vg0 = vg + ((size_t)h * HD + srow) * SEQ + scol;   // [d][key]

  // Q fragments (once)
  const bf16* qrow = qg + ((size_t)h * SEQ + (q0 + mrow)) * HD;
  bf16x16 aq[2];
#pragma unroll
  for (int kb2 = 0; kb2 < 2; ++kb2)
    aq[kb2] = load_frag(qrow + kb2 * 32 + hi * 8, qrow + kb2 * 32 + 16 + hi * 8);

  int segq[8];
#pragma unroll
  for (int r = 0; r < 8; ++r) {
    const int sq = q0 + hi * 8 + r;
    segq[r] = (sq >= o1) + (sq >= o2) + (sq >= o3);
  }

  float mrun[8], lrun[8];
#pragma unroll
  for (int r = 0; r < 8; ++r) { mrun[r] = -1e30f; lrun[r] = 0.f; }
  f32x8 acco[4] = {};
  bf16* pbase = &Ps[wv * 16 * 64];

  // preload key-tile 0
  async_ld_b128(&Ks[0][sidx], Kg0);  async_ld_b128(&Ks[0][sidx + 8], Kg0 + 8);
  async_ld_b128(&Vs[0][sidx], Vg0);  async_ld_b128(&Vs[0][sidx + 8], Vg0 + 8);
  wait_async0();
  __syncthreads();

  for (int it = 0; it < SEQ / 64; ++it) {
    const int kt = it * 64;
    const int cur = it & 1, nxt = cur ^ 1;
    const bool more = (it + 1) < (SEQ / 64);
    if (more) {  // DMA next K/V tile while we compute on this one
      Kg0 += (size_t)64 * HD;  // +64 keys (row stride HD)
      Vg0 += 64;               // +64 keys (contiguous)
      async_ld_b128(&Ks[nxt][sidx], Kg0);  async_ld_b128(&Ks[nxt][sidx + 8], Kg0 + 8);
      async_ld_b128(&Vs[nxt][sidx], Vg0);  async_ld_b128(&Vs[nxt][sidx + 8], Vg0 + 8);
    }
    // ---- S tile = Q . K^T (16x64) ----
    f32x8 sacc[4] = {};
#pragma unroll
    for (int kb2 = 0; kb2 < 2; ++kb2) {
      bf16x16 bk[4];
#pragma unroll
      for (int t = 0; t < 4; ++t) {
        const int row = (t * 16 + mrow) * 64 + kb2 * 32;
        bk[t] = load_frag(&Ks[cur][row + hi * 8], &Ks[cur][row + 16 + hi * 8]);
      }
#pragma unroll
      for (int t = 0; t < 4; ++t)
        sacc[t] = __builtin_amdgcn_wmma_f32_16x16x32_bf16(
            false, aq[kb2], false, bk[t], (short)0, sacc[t], false, false);
    }
    // ---- scale + segment bias + per-row running max ----
    float nm[8];
#pragma unroll
    for (int r = 0; r < 8; ++r) nm[r] = mrun[r];
#pragma unroll
    for (int t = 0; t < 4; ++t) {
      const int key = kt + t * 16 + mrow;
      const int segk = (key >= o1) + (key >= o2) + (key >= o3);
#pragma unroll
      for (int r = 0; r < 8; ++r) {
        float sv = sacc[t][r] * 0.125f + ((segk == segq[r]) ? 1.0f : 0.0f);
        sacc[t][r] = sv;
        nm[r] = fmaxf(nm[r], sv);
      }
    }
#pragma unroll
    for (int msk = 1; msk < 16; msk <<= 1)
#pragma unroll
      for (int r = 0; r < 8; ++r) nm[r] = fmaxf(nm[r], __shfl_xor(nm[r], msk, 32));
#pragma unroll
    for (int r = 0; r < 8; ++r) {
      const float alpha = __expf(mrun[r] - nm[r]);
      mrun[r] = nm[r];
      lrun[r] *= alpha;
#pragma unroll
      for (int t = 0; t < 4; ++t) acco[t][r] *= alpha;
    }
    // ---- P = exp(S - m), partial row sums, write P patch ----
#pragma unroll
    for (int t = 0; t < 4; ++t)
#pragma unroll
      for (int r = 0; r < 8; ++r) {
        const float p = __expf(sacc[t][r] - mrun[r]);
        lrun[r] += p;
        pbase[(r + hi * 8) * 64 + t * 16 + mrow] = (bf16)p;
      }
    asm volatile("s_wait_dscnt 0x0" ::: "memory");  // drain P stores (compiler barrier)
    // ---- O += P @ V ----
#pragma unroll
    for (int kb2 = 0; kb2 < 2; ++kb2) {
      bf16x16 ap = load_frag(pbase + mrow * 64 + kb2 * 32 + hi * 8,
                             pbase + mrow * 64 + kb2 * 32 + 16 + hi * 8);
      bf16x16 bv[4];
#pragma unroll
      for (int t = 0; t < 4; ++t) {
        const int row = (t * 16 + mrow) * 64 + kb2 * 32;
        bv[t] = load_frag(&Vs[cur][row + hi * 8], &Vs[cur][row + 16 + hi * 8]);
      }
#pragma unroll
      for (int t = 0; t < 4; ++t)
        acco[t] = __builtin_amdgcn_wmma_f32_16x16x32_bf16(
            false, ap, false, bv[t], (short)0, acco[t], false, false);
    }
    if (more) wait_async0();
    __syncthreads();
  }
  // ---- finalize: full row sums, normalize, store bf16 ----
#pragma unroll
  for (int msk = 1; msk < 16; msk <<= 1)
#pragma unroll
    for (int r = 0; r < 8; ++r) lrun[r] += __shfl_xor(lrun[r], msk, 32);
#pragma unroll
  for (int t = 0; t < 4; ++t)
#pragma unroll
    for (int r = 0; r < 8; ++r) {
      const float v = acco[t][r] / lrun[r];
      og[(size_t)(q0 + hi * 8 + r) * HID + h * HD + t * 16 + mrow] = (bf16)v;
    }
}

// ---------------------------------------------------------------------------
extern "C" void kernel_launch(void* const* d_in, const int* in_sizes, int n_in,
                              void* d_out, int out_size, void* d_ws, size_t ws_size,
                              hipStream_t stream) {
  (void)in_sizes; (void)n_in; (void)out_size; (void)ws_size;
  const float* x     = (const float*)d_in[0];
  const float* ropeC = (const float*)d_in[1];
  const float* ropeS = (const float*)d_in[2];
  const float* n0w   = (const float*)d_in[3];
  const float* n0b   = (const float*)d_in[4];
  const float* n1w   = (const float*)d_in[5];
  const float* n1b   = (const float*)d_in[6];
  const float* wqkv  = (const float*)d_in[7];
  const float* bqkv  = (const float*)d_in[8];
  const float* wo    = (const float*)d_in[9];
  const float* bo    = (const float*)d_in[10];
  const float* wup   = (const float*)d_in[11];
  const float* bup   = (const float*)d_in[12];
  const float* wdn   = (const float*)d_in[13];
  const float* bdn   = (const float*)d_in[14];
  const int*   offs  = (const int*)d_in[15];

  uint8_t* ws = (uint8_t*)d_ws;
  const size_t MB = 1ull << 20;
  bf16*  wqkv_bf = (bf16*)(ws + 0);        // 6 MB
  bf16*  wo_bf   = (bf16*)(ws + 6 * MB);   // 2 MB
  bf16*  wup_bf  = (bf16*)(ws + 8 * MB);   // 8 MB
  bf16*  wdn_bf  = (bf16*)(ws + 16 * MB);  // 8 MB
  bf16*  h_bf    = (bf16*)(ws + 24 * MB);  // 4 MB (LN0 out, reused for LN1 out)
  float* qkv_f   = (float*)(ws + 28 * MB); // 24 MB (reused as mlp_bf, 16 MB)
  bf16*  mlp_bf  = (bf16*)(ws + 28 * MB);
  bf16*  q_bf    = (bf16*)(ws + 52 * MB);  // 4 MB
  bf16*  k_bf    = (bf16*)(ws + 56 * MB);  // 4 MB
  bf16*  vt_bf   = (bf16*)(ws + 60 * MB);  // 4 MB  [h][d][s]
  bf16*  o_bf    = (bf16*)(ws + 64 * MB);  // 4 MB
  float* x1_f    = (float*)(ws + 68 * MB); // 8 MB residual stream

  cvt_f32_bf16_kernel<<<3072 * 1024 / 1024, 256, 0, stream>>>(wqkv, wqkv_bf, 3072 * 1024);
  cvt_f32_bf16_kernel<<<1024 * 1024 / 1024, 256, 0, stream>>>(wo,   wo_bf,   1024 * 1024);
  cvt_f32_bf16_kernel<<<4096 * 1024 / 1024, 256, 0, stream>>>(wup,  wup_bf,  4096 * 1024);
  cvt_f32_bf16_kernel<<<4096 * 1024 / 1024, 256, 0, stream>>>(wdn,  wdn_bf,  4096 * 1024);

  layernorm_bf16_kernel<<<SEQ, 256, 0, stream>>>(x, n0w, n0b, h_bf);
  gemm_bf16_kernel<<<dim3(3072 / 128, SEQ / 128), 256, 0, stream>>>(
      h_bf, wqkv_bf, bqkv, nullptr, qkv_f, nullptr, SEQ, 3072, HID, 0);
  rope_split_kernel<<<SEQ * NH * 32 / 256, 256, 0, stream>>>(
      qkv_f, ropeC, ropeS, q_bf, k_bf, vt_bf);
  flash_attn_kernel<<<dim3(SEQ / 128, NH), 256, 0, stream>>>(
      q_bf, k_bf, vt_bf, o_bf, offs);
  gemm_bf16_kernel<<<dim3(HID / 128, SEQ / 128), 256, 0, stream>>>(
      o_bf, wo_bf, bo, x, x1_f, nullptr, SEQ, HID, HID, 2);
  layernorm_bf16_kernel<<<SEQ, 256, 0, stream>>>(x1_f, n1w, n1b, h_bf);
  gemm_bf16_kernel<<<dim3(MLPD / 128, SEQ / 128), 256, 0, stream>>>(
      h_bf, wup_bf, bup, nullptr, nullptr, mlp_bf, SEQ, MLPD, HID, 1);
  gemm_bf16_kernel<<<dim3(HID / 128, SEQ / 128), 256, 0, stream>>>(
      mlp_bf, wdn_bf, bdn, x1_f, (float*)d_out, nullptr, SEQ, HID, MLPD, 2);
}
